// Transformer_43593918054955
// MI455X (gfx1250) — compile-verified, matrix operand order checked by reading, not run on hardware
//
#include <hip/hip_runtime.h>
#include <hip/hip_bf16.h>

typedef _Float16 f16;
typedef __attribute__((ext_vector_type(8)))  _Float16 v8h;
typedef __attribute__((ext_vector_type(16))) _Float16 v16h;
typedef __attribute__((ext_vector_type(8)))  float    v8f;

#define BB 8
#define SS 512
#define EE 384
#define HH 6
#define DD 64
#define FFD 1536
#define VV 32000
#define MR (BB*SS)

// Load one 16x32 f16 WMMA fragment (A row r / B column r), per ISA lane layout:
// lanes 0-15 hold K 0..7 & 16..23, lanes 16-31 hold K 8..15 & 24..31.
__device__ __forceinline__ v16h ldfrag(const f16* __restrict__ p, int k0, int koff)
{
    v8h x0 = *(const v8h*)(p + k0 + koff);
    v8h x1 = *(const v8h*)(p + k0 + 16 + koff);
    return __builtin_shufflevector(x0, x1, 0,1,2,3,4,5,6,7,8,9,10,11,12,13,14,15);
}

__device__ __forceinline__ v8f wmma_f16(v16h a, v16h b, v8f c)
{
    return __builtin_amdgcn_wmma_f32_16x16x32_f16(false, a, false, b,
                                                  (short)0, c, false, false);
}

// ---------------------------------------------------------------------------
// Weight fp32 (K x N) -> f16 transposed (N x K)
// ---------------------------------------------------------------------------
__global__ void w_to_f16t(const float* __restrict__ W, f16* __restrict__ Wt,
                          long n, int Nd, int Kd)
{
    long i = (long)blockIdx.x * 256 + threadIdx.x;
    if (i >= n) return;
    long k = i / Nd;
    long c = i % Nd;
    Wt[c * (long)Kd + k] = (f16)W[i];
}

// ---------------------------------------------------------------------------
// Embedding lookup + sinusoidal positional encoding -> f32 + f16
// ---------------------------------------------------------------------------
__global__ void embed_pos(const int* __restrict__ tok, const float* __restrict__ tab,
                          float* __restrict__ o32, f16* __restrict__ o16)
{
    long i = (long)blockIdx.x * 256 + threadIdx.x;
    if (i >= (long)BB * SS * EE) return;
    int e = (int)(i % EE);
    long r = i / EE;
    int s = (int)(r % SS);
    int b = (int)(r / SS);
    float inv = __powf(10000.0f, -2.0f * (float)e / (float)EE);
    float ang = (float)s * inv;
    float pe = (e & 1) ? __cosf(ang) : __sinf(ang);
    int t = tok[b * SS + s];
    float v = tab[(long)t * EE + e] + pe;
    o32[i] = v;
    o16[i] = (f16)v;
}

// ---------------------------------------------------------------------------
// (B,S,H*D) fp32 q/k/v -> qh,kh (B,H,S,D) f16 and vt (B,H,D,S) f16
// ---------------------------------------------------------------------------
__global__ void pack_qkv(const float* __restrict__ q32, const float* __restrict__ k32,
                         const float* __restrict__ v32, f16* __restrict__ qh,
                         f16* __restrict__ kh, f16* __restrict__ vt)
{
    long i = (long)blockIdx.x * 256 + threadIdx.x;
    if (i >= (long)BB * SS * HH * DD) return;
    int d = (int)(i % DD);
    long r = i / DD;
    int h = (int)(r % HH);
    long r2 = r / HH;
    int s = (int)(r2 % SS);
    int b = (int)(r2 / SS);
    long bh = (long)b * HH + h;
    long dstq = (bh * SS + s) * DD + d;
    long dstv = (bh * DD + d) * SS + s;
    qh[dstq] = (f16)q32[i];
    kh[dstq] = (f16)k32[i];
    vt[dstv] = (f16)v32[i];
}

// ---------------------------------------------------------------------------
// Generic batched WMMA GEMM:  C[M x N] = A[M x K](f16) * Bt[N x K](f16)^T + bias
// One wave computes a 32x32 C tile as 2x2 register-blocked 16x16 WMMA tiles:
// 4 v_wmma per K=32 step against 8 b128 loads (2x the compute density of 1x1).
// C batch base = (batch/cInner)*sCo + (batch%cInner)*sCi  (handles (B,S,H,D) C).
// ---------------------------------------------------------------------------
__global__ void __launch_bounds__(256)
gemm_wmma_f16(const f16* __restrict__ A, const f16* __restrict__ Bt,
              const float* __restrict__ bias, float* __restrict__ C,
              f16* __restrict__ C16,
              int M, int N, int K, int lda, int ldb, int ldc,
              long sA, long sB, long sCo, long sCi, int cInner)
{
    int wave = threadIdx.x >> 5;
    int lane = threadIdx.x & 31;
    int tilesX = N >> 5;
    int tilesTotal = (M >> 5) * tilesX;
    int tile = blockIdx.x * 8 + wave;
    if (tile >= tilesTotal) return;
    int tm = tile / tilesX;
    int tn = tile - tm * tilesX;
    int batch = blockIdx.y;

    int r = lane & 15;
    int hi = lane >> 4;
    int koff = hi * 8;

    const f16* Ar0 = A + (long)batch * sA + ((long)tm * 32 + r) * lda;
    const f16* Ar1 = Ar0 + 16 * (long)lda;
    const f16* Br0 = Bt + (long)batch * sB + ((long)tn * 32 + r) * ldb;
    const f16* Br1 = Br0 + 16 * (long)ldb;

    v8f acc00 = {}, acc01 = {}, acc10 = {}, acc11 = {};
    for (int k0 = 0; k0 < K; k0 += 32) {
        v16h a0 = ldfrag(Ar0, k0, koff);
        v16h a1 = ldfrag(Ar1, k0, koff);
        v16h b0 = ldfrag(Br0, k0, koff);
        v16h b1 = ldfrag(Br1, k0, koff);
        acc00 = wmma_f16(a0, b0, acc00);
        acc01 = wmma_f16(a0, b1, acc01);
        acc10 = wmma_f16(a1, b0, acc10);
        acc11 = wmma_f16(a1, b1, acc11);
    }

    long cbase = (long)(batch / cInner) * sCo + (long)(batch % cInner) * sCi
               + (long)tm * 32 * ldc + (long)tn * 32;
    float bv0 = bias ? bias[tn * 32 + r] : 0.0f;
    float bv1 = bias ? bias[tn * 32 + 16 + r] : 0.0f;
    int mrow = hi * 8;
    #pragma unroll
    for (int i = 0; i < 8; ++i) {
        long off0 = cbase + (long)(mrow + i) * ldc + r;        // rows 0..15 of tile
        long off1 = cbase + (long)(mrow + i + 16) * ldc + r;   // rows 16..31
        float v00 = acc00[i] + bv0;
        float v01 = acc01[i] + bv1;
        float v10 = acc10[i] + bv0;
        float v11 = acc11[i] + bv1;
        C[off0]      = v00;
        C[off0 + 16] = v01;
        C[off1]      = v10;
        C[off1 + 16] = v11;
        if (C16) {
            C16[off0]      = (f16)v00;
            C16[off0 + 16] = (f16)v01;
            C16[off1]      = (f16)v10;
            C16[off1 + 16] = (f16)v11;
        }
    }
}

// ---------------------------------------------------------------------------
// Masked softmax over one row of S=512 scores; mask from key tokens (+causal).
// scores were accumulated unscaled; apply 1/sqrt(64) here, emit f16 probs.
// ---------------------------------------------------------------------------
__global__ void __launch_bounds__(256)
softmax_mask(const float* __restrict__ S32, f16* __restrict__ P16,
             const int* __restrict__ ktok, int causal, float scale)
{
    int row = blockIdx.x;                 // (b*H + h)*S + q
    int q = row % SS;
    int b = row / (HH * SS);
    const float* src = S32 + (long)row * SS;
    f16* dst = P16 + (long)row * SS;
    __shared__ float red[256];

    float vals[2];
    float mx = -3.4e38f;
    #pragma unroll
    for (int t = 0; t < 2; ++t) {
        int k = threadIdx.x + t * 256;
        float v = src[k] * scale;
        if (ktok[b * SS + k] == 0) v += -1e9f;
        if (causal && k > q) v += -1e9f;
        vals[t] = v;
        mx = fmaxf(mx, v);
    }
    red[threadIdx.x] = mx; __syncthreads();
    for (int s = 128; s > 0; s >>= 1) {
        if (threadIdx.x < (unsigned)s) red[threadIdx.x] = fmaxf(red[threadIdx.x], red[threadIdx.x + s]);
        __syncthreads();
    }
    mx = red[0]; __syncthreads();

    float sum = 0.0f;
    #pragma unroll
    for (int t = 0; t < 2; ++t) { vals[t] = __expf(vals[t] - mx); sum += vals[t]; }
    red[threadIdx.x] = sum; __syncthreads();
    for (int s = 128; s > 0; s >>= 1) {
        if (threadIdx.x < (unsigned)s) red[threadIdx.x] += red[threadIdx.x + s];
        __syncthreads();
    }
    float inv = 1.0f / red[0];
    #pragma unroll
    for (int t = 0; t < 2; ++t)
        dst[threadIdx.x + t * 256] = (f16)(vals[t] * inv);
}

// ---------------------------------------------------------------------------
// out = LayerNorm(x + delta) * g + b  -> f32 stream (in place) + f16 mirror
// ---------------------------------------------------------------------------
__global__ void __launch_bounds__(128)
add_ln(const float* __restrict__ X, const float* __restrict__ D,
       const float* __restrict__ g, const float* __restrict__ b,
       float* __restrict__ O32, f16* __restrict__ O16)
{
    long base = (long)blockIdx.x * EE;
    __shared__ float red[128];
    float v[3];
    float s = 0.0f;
    #pragma unroll
    for (int t = 0; t < 3; ++t) {
        int i = threadIdx.x + t * 128;
        v[t] = X[base + i] + D[base + i];
        s += v[t];
    }
    red[threadIdx.x] = s; __syncthreads();
    for (int st = 64; st > 0; st >>= 1) {
        if (threadIdx.x < (unsigned)st) red[threadIdx.x] += red[threadIdx.x + st];
        __syncthreads();
    }
    float mean = red[0] / (float)EE; __syncthreads();
    float s2 = 0.0f;
    #pragma unroll
    for (int t = 0; t < 3; ++t) { float d0 = v[t] - mean; s2 += d0 * d0; }
    red[threadIdx.x] = s2; __syncthreads();
    for (int st = 64; st > 0; st >>= 1) {
        if (threadIdx.x < (unsigned)st) red[threadIdx.x] += red[threadIdx.x + st];
        __syncthreads();
    }
    float rstd = rsqrtf(red[0] / (float)EE + 1e-5f);
    #pragma unroll
    for (int t = 0; t < 3; ++t) {
        int i = threadIdx.x + t * 128;
        float o = (v[t] - mean) * rstd * g[i] + b[i];
        O32[base + i] = o;
        O16[base + i] = (f16)o;
    }
}

// ---------------------------------------------------------------------------
// Fused vocab projection + argmax. Block = 8 waves sharing 32 rows; each wave
// sweeps 32-col tiles (stride 8) of the 32000-wide logits: 2x2 register-blocked
// WMMA (48 v_wmma per tile, K=384), tracking per-lane argmax. Softmax is
// monotonic so argmax(logits) == reference argmax.
// ---------------------------------------------------------------------------
__global__ void __launch_bounds__(256)
vocab_argmax(const f16* __restrict__ A, const f16* __restrict__ Wt,
             const float* __restrict__ bias, float* __restrict__ out)
{
    int wave = threadIdx.x >> 5;
    int lane = threadIdx.x & 31;
    int r = lane & 15;
    int hi = lane >> 4;
    int koff = hi * 8;
    long rowBase = (long)blockIdx.x * 32;
    const f16* Ar0 = A + (rowBase + r) * EE;
    const f16* Ar1 = A + (rowBase + 16 + r) * EE;

    float bestV[2][8];
    int   bestI[2][8];
    #pragma unroll
    for (int rt = 0; rt < 2; ++rt)
        #pragma unroll
        for (int i = 0; i < 8; ++i) { bestV[rt][i] = -3.4e38f; bestI[rt][i] = 0; }

    for (int tn = wave; tn < (VV >> 5); tn += 8) {
        const f16* Br0 = Wt + (long)(tn * 32 + r) * EE;
        const f16* Br1 = Br0 + 16 * (long)EE;
        v8f acc00 = {}, acc01 = {}, acc10 = {}, acc11 = {};
        #pragma unroll
        for (int k0 = 0; k0 < EE; k0 += 32) {
            v16h a0 = ldfrag(Ar0, k0, koff);
            v16h a1 = ldfrag(Ar1, k0, koff);
            v16h b0 = ldfrag(Br0, k0, koff);
            v16h b1 = ldfrag(Br1, k0, koff);
            acc00 = wmma_f16(a0, b0, acc00);
            acc01 = wmma_f16(a0, b1, acc01);
            acc10 = wmma_f16(a1, b0, acc10);
            acc11 = wmma_f16(a1, b1, acc11);
        }
        int c0 = tn * 32 + r;
        int c1 = c0 + 16;
        float bv0 = bias[c0];
        float bv1 = bias[c1];
        #pragma unroll
        for (int i = 0; i < 8; ++i) {
            float v;
            v = acc00[i] + bv0;
            if (v > bestV[0][i]) { bestV[0][i] = v; bestI[0][i] = c0; }
            v = acc01[i] + bv1;
            if (v > bestV[0][i]) { bestV[0][i] = v; bestI[0][i] = c1; }
            v = acc10[i] + bv0;
            if (v > bestV[1][i]) { bestV[1][i] = v; bestI[1][i] = c0; }
            v = acc11[i] + bv1;
            if (v > bestV[1][i]) { bestV[1][i] = v; bestI[1][i] = c1; }
        }
    }

    __shared__ float sv[32 * 128];
    __shared__ int   si[32 * 128];
    #pragma unroll
    for (int rt = 0; rt < 2; ++rt)
        #pragma unroll
        for (int i = 0; i < 8; ++i) {
            int row = rt * 16 + hi * 8 + i;
            int slot = row * 128 + wave * 16 + r;
            sv[slot] = bestV[rt][i];
            si[slot] = bestI[rt][i];
        }
    __syncthreads();
    if (threadIdx.x < 32) {
        int row = threadIdx.x;
        float bv2 = -3.4e38f;
        int bi = 0x7fffffff;
        for (int j = 0; j < 128; ++j) {
            float v = sv[row * 128 + j];
            int ix = si[row * 128 + j];
            if (v > bv2 || (v == bv2 && ix < bi)) { bv2 = v; bi = ix; }
        }
        out[rowBase + row] = (float)bi;
    }
}

// ---------------------------------------------------------------------------
// Host orchestration
// ---------------------------------------------------------------------------
extern "C" void kernel_launch(void* const* d_in, const int* in_sizes, int n_in,
                              void* d_out, int out_size, void* d_ws, size_t ws_size,
                              hipStream_t stream)
{
    (void)in_sizes; (void)n_in; (void)out_size; (void)ws_size;

    const int*   x     = (const int*)d_in[0];
    const int*   y     = (const int*)d_in[1];
    const float* x_emb = (const float*)d_in[2];
    const float* y_emb = (const float*)d_in[3];
    int idx = 4;
    const float* enc[6][16];
    for (int l = 0; l < 6; ++l)
        for (int t = 0; t < 16; ++t) enc[l][t] = (const float*)d_in[idx++];
    const float* dec[6][26];
    for (int l = 0; l < 6; ++l)
        for (int t = 0; t < 26; ++t) dec[l][t] = (const float*)d_in[idx++];
    const float* out_w = (const float*)d_in[idx++];
    const float* out_b = (const float*)d_in[idx++];

    // ---- workspace bump allocator ----
    char* wsb = (char*)d_ws;
    size_t off = 0;
    auto alloc = [&](size_t bytes) -> void* {
        void* p = wsb + off;
        off += (bytes + 255) & ~(size_t)255;
        return p;
    };

    const size_t ATT_W = (size_t)EE * EE * sizeof(f16);
    const size_t FF_W  = (size_t)EE * FFD * sizeof(f16);

    f16* encW[6][6];   // wqT,wkT,wvT,woT,w1T,w2T
    for (int l = 0; l < 6; ++l) {
        for (int t = 0; t < 4; ++t) encW[l][t] = (f16*)alloc(ATT_W);
        encW[l][4] = (f16*)alloc(FF_W);
        encW[l][5] = (f16*)alloc(FF_W);
    }
    f16* decW[6][10];  // s:wqT..woT, c:wqT..woT, w1T, w2T
    for (int l = 0; l < 6; ++l) {
        for (int t = 0; t < 8; ++t) decW[l][t] = (f16*)alloc(ATT_W);
        decW[l][8] = (f16*)alloc(FF_W);
        decW[l][9] = (f16*)alloc(FF_W);
    }
    f16* outwT = (f16*)alloc((size_t)EE * VV * sizeof(f16));

    const size_t BSE32 = (size_t)MR * EE * sizeof(float);
    const size_t BSE16 = (size_t)MR * EE * sizeof(f16);
    const size_t HEAD16 = (size_t)BB * HH * SS * DD * sizeof(f16);

    float* henc32 = (float*)alloc(BSE32);
    f16*   henc16 = (f16*)alloc(BSE16);
    float* hdec32 = (float*)alloc(BSE32);
    f16*   hdec16 = (f16*)alloc(BSE16);
    float* q32 = (float*)alloc(BSE32);
    float* k32 = (float*)alloc(BSE32);
    float* v32 = (float*)alloc(BSE32);
    f16*   qh  = (f16*)alloc(HEAD16);
    f16*   kh  = (f16*)alloc(HEAD16);
    f16*   vt  = (f16*)alloc(HEAD16);
    float* scores32 = (float*)alloc((size_t)BB * HH * SS * SS * sizeof(float));
    f16*   probs16  = (f16*)alloc((size_t)BB * HH * SS * SS * sizeof(f16));
    float* ao32 = (float*)alloc(BSE32);
    f16*   ao16 = (f16*)alloc(BSE16);
    float* proj32 = (float*)alloc(BSE32);
    float* ff32 = (float*)alloc((size_t)MR * FFD * sizeof(float));
    f16*   ff16 = (f16*)alloc((size_t)MR * FFD * sizeof(f16));

    // ---- weight conversion (f32 KxN -> f16 NxK) ----
    auto conv = [&](const float* W, f16* Wt, int Kd, int Nd) {
        long n = (long)Kd * Nd;
        w_to_f16t<<<dim3((unsigned)((n + 255) / 256)), 256, 0, stream>>>(W, Wt, n, Nd, Kd);
    };
    for (int l = 0; l < 6; ++l) {
        conv(enc[l][0], encW[l][0], EE, EE);   // wq
        conv(enc[l][2], encW[l][1], EE, EE);   // wk
        conv(enc[l][4], encW[l][2], EE, EE);   // wv
        conv(enc[l][6], encW[l][3], EE, EE);   // wo
        conv(enc[l][12], encW[l][4], EE, FFD); // w1
        conv(enc[l][14], encW[l][5], FFD, EE); // w2
    }
    for (int l = 0; l < 6; ++l) {
        conv(dec[l][0],  decW[l][0], EE, EE);
        conv(dec[l][2],  decW[l][1], EE, EE);
        conv(dec[l][4],  decW[l][2], EE, EE);
        conv(dec[l][6],  decW[l][3], EE, EE);
        conv(dec[l][8],  decW[l][4], EE, EE);
        conv(dec[l][10], decW[l][5], EE, EE);
        conv(dec[l][12], decW[l][6], EE, EE);
        conv(dec[l][14], decW[l][7], EE, EE);
        conv(dec[l][22], decW[l][8], EE, FFD);
        conv(dec[l][24], decW[l][9], FFD, EE);
    }
    conv(out_w, outwT, EE, VV);

    // ---- launch helpers ----
    auto gemm = [&](const f16* A, const f16* Bt, const float* bias, float* C, f16* C16,
                    int M, int N, int K, int lda, int ldb, int ldc,
                    int batch, long sA, long sB, long sCo, long sCi, int cInner) {
        int tiles = (M / 32) * (N / 32);
        dim3 g((unsigned)((tiles + 7) / 8), (unsigned)batch);
        gemm_wmma_f16<<<g, 256, 0, stream>>>(A, Bt, bias, C, C16,
                                             M, N, K, lda, ldb, ldc,
                                             sA, sB, sCo, sCi, cInner);
    };

    auto attention = [&](const f16* q_in16, const f16* kv_in16, const int* ktok, int causal,
                         f16* wqT, const float* bq, f16* wkT, const float* bk,
                         f16* wvT, const float* bv, f16* woT, const float* bo,
                         float* proj_out) {
        gemm(q_in16, wqT, bq, q32, nullptr, MR, EE, EE, EE, EE, EE, 1, 0, 0, 0, 0, 1);
        gemm(kv_in16, wkT, bk, k32, nullptr, MR, EE, EE, EE, EE, EE, 1, 0, 0, 0, 0, 1);
        gemm(kv_in16, wvT, bv, v32, nullptr, MR, EE, EE, EE, EE, EE, 1, 0, 0, 0, 0, 1);
        long nqkv = (long)BB * SS * HH * DD;
        pack_qkv<<<dim3((unsigned)((nqkv + 255) / 256)), 256, 0, stream>>>(q32, k32, v32, qh, kh, vt);
        // scores[bh] = Q[bh] (SxD) * K[bh]^T  -> (SxS)
        gemm(qh, kh, nullptr, scores32, nullptr, SS, SS, DD, DD, DD, SS,
             BB * HH, (long)SS * DD, (long)SS * DD, (long)SS * SS, 0, 1);
        softmax_mask<<<dim3((unsigned)(BB * HH * SS)), 256, 0, stream>>>(
            scores32, probs16, ktok, causal, 0.125f);
        // out[bh] = P[bh] (SxS) * V[bh] (SxD), written into (B,S,H,D)
        gemm(probs16, vt, nullptr, ao32, ao16, SS, DD, SS, SS, SS, HH * DD,
             BB * HH, (long)SS * SS, (long)DD * SS, (long)SS * HH * DD, DD, HH);
        gemm(ao16, woT, bo, proj_out, nullptr, MR, EE, EE, EE, EE, EE, 1, 0, 0, 0, 0, 1);
    };

    auto ffn = [&](const f16* in16, f16* w1T, const float* b1, f16* w2T, const float* b2,
                   float* out32) {
        gemm(in16, w1T, b1, ff32, ff16, MR, FFD, EE, EE, EE, FFD, 1, 0, 0, 0, 0, 1);
        gemm(ff16, w2T, b2, out32, nullptr, MR, EE, FFD, FFD, FFD, EE, 1, 0, 0, 0, 0, 1);
    };

    auto lnres = [&](float* h32s, f16* h16s, const float* delta,
                     const float* g, const float* bb) {
        add_ln<<<dim3((unsigned)MR), 128, 0, stream>>>(h32s, delta, g, bb, h32s, h16s);
    };

    // ---- embeddings ----
    long nemb = (long)MR * EE;
    embed_pos<<<dim3((unsigned)((nemb + 255) / 256)), 256, 0, stream>>>(x, x_emb, henc32, henc16);
    embed_pos<<<dim3((unsigned)((nemb + 255) / 256)), 256, 0, stream>>>(y, y_emb, hdec32, hdec16);

    // ---- encoder ----
    for (int l = 0; l < 6; ++l) {
        attention(henc16, henc16, x, 0,
                  encW[l][0], enc[l][1], encW[l][1], enc[l][3],
                  encW[l][2], enc[l][5], encW[l][3], enc[l][7], proj32);
        lnres(henc32, henc16, proj32, enc[l][8], enc[l][9]);
        ffn(henc16, encW[l][4], enc[l][13], encW[l][5], enc[l][15], proj32);
        lnres(henc32, henc16, proj32, enc[l][10], enc[l][11]);
    }

    // ---- decoder ----
    for (int l = 0; l < 6; ++l) {
        attention(hdec16, hdec16, y, 1,
                  decW[l][0], dec[l][1], decW[l][1], dec[l][3],
                  decW[l][2], dec[l][5], decW[l][3], dec[l][7], proj32);
        lnres(hdec32, hdec16, proj32, dec[l][16], dec[l][17]);
        attention(hdec16, henc16, x, 0,
                  decW[l][4], dec[l][9], decW[l][5], dec[l][11],
                  decW[l][6], dec[l][13], decW[l][7], dec[l][15], proj32);
        lnres(hdec32, hdec16, proj32, dec[l][18], dec[l][19]);
        ffn(hdec16, decW[l][8], dec[l][23], decW[l][9], dec[l][25], proj32);
        lnres(hdec32, hdec16, proj32, dec[l][20], dec[l][21]);
    }

    // ---- fused vocab projection + argmax ----
    vocab_argmax<<<dim3((unsigned)(MR / 32)), 256, 0, stream>>>(
        hdec16, outwT, out_b, (float*)d_out);
}
